// GATModel_9294309228705
// MI455X (gfx1250) — matrix operand order repaired and checked by reading, not compiled, not run on hardware
//
#include <hip/hip_runtime.h>
#include <hip/hip_bf16.h>

#define NEG_SLOPE 0.2f

typedef __attribute__((ext_vector_type(16))) __bf16 v16bf;
typedef __attribute__((ext_vector_type(8)))  float  v8f;

union Frag { v16bf v; uint4 u[2]; };

// ---------------- edge list with self loops ----------------
__global__ void k_build_edges(const int* __restrict__ ei, int* __restrict__ src,
                              int* __restrict__ dst, int nE, int nN) {
  int i = blockIdx.x * blockDim.x + threadIdx.x;
  if (i < nE) {
    src[i] = ei[i];
    dst[i] = ei[nE + i];
  } else if (i < nE + nN) {
    int n = i - nE;
    src[i] = n;
    dst[i] = n;
  }
}

// ---------------- f32 -> bf16 convert ----------------
__global__ void k_f32_to_bf16(const float* __restrict__ in, __bf16* __restrict__ out, int n) {
  int i = blockIdx.x * blockDim.x + threadIdx.x;
  if (i < n) out[i] = (__bf16)in[i];
}

// W [K,N] f32 row-major -> Wt [N,K] bf16 (so B-fragment loads are contiguous)
__global__ void k_transpose_bf16(const float* __restrict__ W, __bf16* __restrict__ Wt,
                                 int K, int N) {
  int i = blockIdx.x * blockDim.x + threadIdx.x;
  if (i < K * N) {
    int k = i / N, n = i - k * N;
    Wt[(size_t)n * K + k] = (__bf16)W[i];
  }
}

// ---------------- WMMA GEMM: C[M,N] = A[M,K] * Bt[N,K]^T ----------------
// One wave per 16x64 output strip (4 WMMA tiles sharing one A fragment).
// M % 16 == 0, N % 64 == 0, K % 32 == 0 (true for all three layers).
__global__ void k_gemm_wmma(const __bf16* __restrict__ A, const __bf16* __restrict__ Bt,
                            float* __restrict__ C, int K, int N, int mtiles, int ngroups) {
  int wid = blockIdx.x * (blockDim.x >> 5) + (threadIdx.x >> 5);
  if (wid >= mtiles * ngroups) return;  // wave-uniform: EXEC stays all-ones for live waves
  int tm = wid / ngroups, tg = wid - tm * ngroups;
  int lane = threadIdx.x & 31;
  int l15 = lane & 15, hi = lane >> 4;

  // A 16x32 bf16 fragment: lanes 0-15 row=l15 K=0..15, lanes 16-31 same rows
  // K=16..31. Per lane: one contiguous 32B run (two 16B loads).
  const __bf16* Ap = A + (size_t)(tm * 16 + l15) * K + hi * 16;
  // B 32x16 bf16 fragment: lane = col, lanes 0-15 K=0..15, lanes 16-31 K=16..31 -> 32B run.
  const __bf16* Bp = Bt + (size_t)(tg * 64 + l15) * K + hi * 16;
  size_t bstride = (size_t)16 * K;  // advance 16 output columns

  v8f acc0 = {0.f, 0.f, 0.f, 0.f, 0.f, 0.f, 0.f, 0.f};
  v8f acc1 = acc0, acc2 = acc0, acc3 = acc0;

  for (int kt = 0; kt < K; kt += 32) {
    __builtin_prefetch((const void*)(Ap + kt + 64), 0, 0);  // next A cacheline
    Frag a, b0, b1, b2, b3;
    a.u[0]  = *(const uint4*)(Ap + kt);
    a.u[1]  = *(const uint4*)(Ap + kt + 8);
    b0.u[0] = *(const uint4*)(Bp + kt);
    b0.u[1] = *(const uint4*)(Bp + kt + 8);
    b1.u[0] = *(const uint4*)(Bp + bstride + kt);
    b1.u[1] = *(const uint4*)(Bp + bstride + kt + 8);
    b2.u[0] = *(const uint4*)(Bp + 2 * bstride + kt);
    b2.u[1] = *(const uint4*)(Bp + 2 * bstride + kt + 8);
    b3.u[0] = *(const uint4*)(Bp + 3 * bstride + kt);
    b3.u[1] = *(const uint4*)(Bp + 3 * bstride + kt + 8);
    acc0 = __builtin_amdgcn_wmma_f32_16x16x32_bf16(false, a.v, false, b0.v,
                                                   (short)0, acc0, false, false);
    acc1 = __builtin_amdgcn_wmma_f32_16x16x32_bf16(false, a.v, false, b1.v,
                                                   (short)0, acc1, false, false);
    acc2 = __builtin_amdgcn_wmma_f32_16x16x32_bf16(false, a.v, false, b2.v,
                                                   (short)0, acc2, false, false);
    acc3 = __builtin_amdgcn_wmma_f32_16x16x32_bf16(false, a.v, false, b3.v,
                                                   (short)0, acc3, false, false);
  }

  // C/D layout: lanes 0-15 col=l15 rows tm*16+v; lanes 16-31 rows tm*16+8+v.
  float* Cp = C + (size_t)(tm * 16 + hi * 8) * N + tg * 64 + l15;
#pragma unroll
  for (int v = 0; v < 8; ++v) {
    float* r = Cp + (size_t)v * N;
    r[0]  = acc0[v];
    r[16] = acc1[v];
    r[32] = acc2[v];
    r[48] = acc3[v];
  }
}

// ---------------- attention logits per (node, head) ----------------
__global__ void k_attn_scores(const float* __restrict__ H, const float* __restrict__ As,
                              const float* __restrict__ Ad, float* __restrict__ asrc,
                              float* __restrict__ adst, int nN, int hs) {
  int i = blockIdx.x * blockDim.x + threadIdx.x;
  if (i >= (nN << hs)) return;
  int h = i & ((1 << hs) - 1);
  const float* hp = H + ((size_t)i << 6);  // i * 64 == (n*heads + h) * 64
  const float* sp = As + h * 64;
  const float* dp = Ad + h * 64;
  float s = 0.f, d = 0.f;
#pragma unroll 8
  for (int c = 0; c < 64; ++c) { float v = hp[c]; s += v * sp[c]; d += v * dp[c]; }
  asrc[i] = s;
  adst[i] = d;
}

// order-preserving float<->uint encoding so segment-max is a u32 atomicMax
// (memset-0 init encodes below every real value)
__device__ __forceinline__ unsigned fenc(float f) {
  unsigned b = __float_as_uint(f);
  return (b & 0x80000000u) ? ~b : (b | 0x80000000u);
}
__device__ __forceinline__ float fdec(unsigned u) {
  unsigned b = (u & 0x80000000u) ? (u ^ 0x80000000u) : ~u;
  return __uint_as_float(b);
}

__global__ void k_edge_max(const float* __restrict__ asrc, const float* __restrict__ adst,
                           const int* __restrict__ src, const int* __restrict__ dst,
                           unsigned* __restrict__ m, int nEt, int hs) {
  int i = blockIdx.x * blockDim.x + threadIdx.x;
  if (i >= (nEt << hs)) return;
  int e = i >> hs, h = i & ((1 << hs) - 1);
  float v = asrc[(src[e] << hs) + h] + adst[(dst[e] << hs) + h];
  v = v > 0.f ? v : NEG_SLOPE * v;
  atomicMax(&m[(dst[e] << hs) + h], fenc(v));
}

__global__ void k_edge_expsum(const float* __restrict__ asrc, const float* __restrict__ adst,
                              const int* __restrict__ src, const int* __restrict__ dst,
                              const unsigned* __restrict__ m, float* __restrict__ ex,
                              float* __restrict__ denom, int nEt, int hs) {
  int i = blockIdx.x * blockDim.x + threadIdx.x;
  if (i >= (nEt << hs)) return;
  int e = i >> hs, h = i & ((1 << hs) - 1);
  int d = dst[e];
  float v = asrc[(src[e] << hs) + h] + adst[(d << hs) + h];
  v = v > 0.f ? v : NEG_SLOPE * v;
  float exv = __expf(v - fdec(m[(d << hs) + h]));
  ex[i] = exv;
  atomicAdd(&denom[(d << hs) + h], exv);
}

// one wave32 per (edge, head): 32 lanes cover C=64 with 2 atomic f32 adds each
__global__ void k_edge_aggregate(const float* __restrict__ H, const int* __restrict__ src,
                                 const int* __restrict__ dst, const float* __restrict__ ex,
                                 const float* __restrict__ denom, float* __restrict__ agg,
                                 int nEt, int hs) {
  int wid = blockIdx.x * (blockDim.x >> 5) + (threadIdx.x >> 5);
  if (wid >= (nEt << hs)) return;
  int e = wid >> hs, h = wid & ((1 << hs) - 1);
  int lane = threadIdx.x & 31;
  int s = src[e], d = dst[e];
  float alpha = ex[wid] / (denom[(d << hs) + h] + 1e-16f);
  const float* hp = H + (((size_t)(s << hs) + h) << 6);
  float* ap = agg + (((size_t)(d << hs) + h) << 6);
  atomicAdd(ap + lane, alpha * hp[lane]);
  atomicAdd(ap + lane + 32, alpha * hp[lane + 32]);
}

__global__ void k_head_mean_bias(const float* __restrict__ agg, const float* __restrict__ bias,
                                 float* __restrict__ out, int nN, int heads) {
  int i = blockIdx.x * blockDim.x + threadIdx.x;
  if (i >= nN * 64) return;
  int n = i >> 6, c = i & 63;
  const float* ap = agg + (size_t)n * heads * 64 + c;
  float s = 0.f;
  for (int h = 0; h < heads; ++h) s += ap[h * 64];
  out[i] = s / (float)heads + bias[c];
}

// ---------------- batchnorm over nodes (64 channels) + relu ----------------
__global__ void k_bn_stats(const float* __restrict__ x, float* __restrict__ sums, int nN) {
  int t = blockIdx.x * blockDim.x + threadIdx.x;
  int c = t & 63;
  int r0 = t >> 6;
  int stride = (gridDim.x * blockDim.x) >> 6;
  float s = 0.f, q = 0.f;
  for (int r = r0; r < nN; r += stride) {
    float v = x[(size_t)r * 64 + c];
    s += v;
    q += v * v;
  }
  atomicAdd(&sums[c], s);
  atomicAdd(&sums[64 + c], q);
}

__global__ void k_bn_relu(const float* __restrict__ x, const float* __restrict__ sums,
                          const float* __restrict__ g, const float* __restrict__ be,
                          float* __restrict__ out, int nN) {
  int i = blockIdx.x * blockDim.x + threadIdx.x;
  if (i >= nN * 64) return;
  int c = i & 63;
  float mean = sums[c] / (float)nN;
  float var = sums[64 + c] / (float)nN - mean * mean;
  float y = (x[i] - mean) * rsqrtf(var + 1e-5f) * g[c] + be[c];
  out[i] = y > 0.f ? y : 0.f;
}

extern "C" void kernel_launch(void* const* d_in, const int* in_sizes, int n_in,
                              void* d_out, int out_size, void* d_ws, size_t ws_size,
                              hipStream_t stream) {
  const int IN_CH = 128, C = 64;
  const int nN = in_sizes[0] / IN_CH;   // 50000 (divisible by 16: 3125 tiles)
  const int nE = in_sizes[1] / 2;       // 800000
  const int nEt = nE + nN;              // + self loops

  const float* x = (const float*)d_in[0];
  const int* ei = (const int*)d_in[1];
  const float* W[3]  = {(const float*)d_in[2],  (const float*)d_in[8],  (const float*)d_in[14]};
  const float* As[3] = {(const float*)d_in[3],  (const float*)d_in[9],  (const float*)d_in[15]};
  const float* Ad[3] = {(const float*)d_in[4],  (const float*)d_in[10], (const float*)d_in[16]};
  const float* Bi[3] = {(const float*)d_in[5],  (const float*)d_in[11], (const float*)d_in[17]};
  const float* Gm[2] = {(const float*)d_in[6],  (const float*)d_in[12]};
  const float* Bb[2] = {(const float*)d_in[7],  (const float*)d_in[13]};
  const int heads[3]  = {8, 4, 4};
  const int hshift[3] = {3, 2, 2};
  const int inch[3]   = {IN_CH, C, C};

  // workspace map (reused across layers)
  char* ws = (char*)d_ws;
  size_t off = 0;
  auto alloc = [&](size_t bytes) -> void* {
    void* p = ws + off;
    off += (bytes + 255) & ~(size_t)255;
    return p;
  };
  int*      srcA = (int*)alloc((size_t)nEt * 4);
  int*      dstA = (int*)alloc((size_t)nEt * 4);
  __bf16*   xbf  = (__bf16*)alloc((size_t)nN * IN_CH * 2);
  __bf16*   wt   = (__bf16*)alloc((size_t)512 * 128 * 2);
  float*    hbuf = (float*)alloc((size_t)nN * 512 * 4);
  float*    agg  = (float*)alloc((size_t)nN * 512 * 4);
  float*    asr  = (float*)alloc((size_t)nN * 8 * 4);
  float*    ads  = (float*)alloc((size_t)nN * 8 * 4);
  unsigned* mbuf = (unsigned*)alloc((size_t)nN * 8 * 4);
  float*    den  = (float*)alloc((size_t)nN * 8 * 4);
  float*    exb  = (float*)alloc((size_t)nEt * 8 * 4);
  float*    l1   = (float*)alloc((size_t)nN * C * 4);
  float*    l2   = (float*)alloc((size_t)nN * C * 4);
  float*    sums = (float*)alloc(128 * 4);
  (void)ws_size; (void)n_in; (void)out_size;

  auto cdiv = [](long long a, long long b) { return (int)((a + b - 1) / b); };

  k_build_edges<<<cdiv(nEt, 256), 256, 0, stream>>>(ei, srcA, dstA, nE, nN);

  const float* lin = x;
  for (int L = 0; L < 3; ++L) {
    int K = inch[L], Hh = heads[L], hs = hshift[L], HC = Hh * C;

    int nx = nN * K;
    k_f32_to_bf16<<<cdiv(nx, 256), 256, 0, stream>>>(lin, xbf, nx);
    k_transpose_bf16<<<cdiv((long long)K * HC, 256), 256, 0, stream>>>(W[L], wt, K, HC);

    int mt = nN / 16, ng = HC / 64;
    long long tiles = (long long)mt * ng;
    k_gemm_wmma<<<cdiv(tiles, 8), 256, 0, stream>>>(xbf, wt, hbuf, K, HC, mt, ng);

    k_attn_scores<<<cdiv((long long)nN * Hh, 256), 256, 0, stream>>>(hbuf, As[L], Ad[L],
                                                                     asr, ads, nN, hs);
    hipMemsetAsync(mbuf, 0, (size_t)nN * Hh * 4, stream);  // 0 == encoded -inf
    hipMemsetAsync(den, 0, (size_t)nN * Hh * 4, stream);
    hipMemsetAsync(agg, 0, (size_t)nN * HC * 4, stream);

    long long eh = (long long)nEt * Hh;
    k_edge_max<<<cdiv(eh, 256), 256, 0, stream>>>(asr, ads, srcA, dstA, mbuf, nEt, hs);
    k_edge_expsum<<<cdiv(eh, 256), 256, 0, stream>>>(asr, ads, srcA, dstA, mbuf, exb, den,
                                                     nEt, hs);
    k_edge_aggregate<<<cdiv(eh, 8), 256, 0, stream>>>(hbuf, srcA, dstA, exb, den, agg,
                                                      nEt, hs);

    float* lout = (L == 2) ? (float*)d_out : ((L == 0) ? l1 : l2);
    k_head_mean_bias<<<cdiv((long long)nN * 64, 256), 256, 0, stream>>>(agg, Bi[L], lout,
                                                                        nN, Hh);
    if (L < 2) {
      hipMemsetAsync(sums, 0, 128 * 4, stream);
      k_bn_stats<<<128, 256, 0, stream>>>(lout, sums, nN);
      k_bn_relu<<<cdiv((long long)nN * 64, 256), 256, 0, stream>>>(lout, sums, Gm[L], Bb[L],
                                                                   lout, nN);
      lin = lout;
    }
  }
}
